// Spa_Module_257698037783
// MI455X (gfx1250) — matrix operand order repaired and checked by reading, not compiled, not run on hardware
//
#include <hip/hip_runtime.h>

typedef __attribute__((ext_vector_type(16))) _Float16 v16h;
typedef __attribute__((ext_vector_type(2)))  __fp16   v2fp;
typedef __attribute__((ext_vector_type(8)))  float    v8f;
typedef __attribute__((ext_vector_type(4)))  unsigned int v4u;

union FragU { v16h v; v4u q[2]; };
union H2U  { _Float16 h; unsigned short u; };
union P2U  { v2fp h; unsigned int u; };

#define C_DIM 64
#define N_DIM 4096
#define QT 64       // queries per block (4 waves x 16)
#define MT 64       // keys per iteration
#define LOG2E 1.4426950408889634f

// Raw v_exp_f32 (no IEEE denorm fixup sequence) — softmax never needs sub-2^-126 weights.
#if __has_builtin(__builtin_amdgcn_exp2f)
#define EXP2F(x) __builtin_amdgcn_exp2f(x)
#else
#define EXP2F(x) __builtin_exp2f(x)
#endif

// A-layout fragment (16x32 f16) from LDS laid out [row][col], rowStride in u16 units.
__device__ inline v16h load_a_frag(const unsigned short* base, int rowStride,
                                   int rowBase, int colBase, int lane) {
  int r  = rowBase + (lane & 15);
  int ch = (lane & 16) ? 8 : 0;
  const unsigned short* p = base + r * rowStride + colBase + ch;
  FragU f;
  f.q[0] = *(const v4u*)(p);
  f.q[1] = *(const v4u*)(p + 16);
  return f.v;
}

// B-layout fragment (32x16 f16) from LDS laid out [col][k], rowStride in u16 units.
__device__ inline v16h load_b_frag(const unsigned short* base, int rowStride,
                                   int colBase, int kBase, int lane) {
  int c  = colBase + (lane & 15);
  int ks = kBase + ((lane & 16) ? 8 : 0);
  const unsigned short* p = base + c * rowStride + ks;
  FragU f;
  f.q[0] = *(const v4u*)(p);
  f.q[1] = *(const v4u*)(p + 16);
  return f.v;
}

#define WMMA_F16(A, B, Cacc) \
  __builtin_amdgcn_wmma_f32_16x16x32_f16(false, (A), false, (B), (short)0, (Cacc), false, false)

// ---- pre-convert x (f32) -> f16 bits in scratch, same [b][c][n] layout ----
__global__ __launch_bounds__(256) void cvt_f16_kernel(const float* __restrict__ x,
                                                      unsigned short* __restrict__ xh,
                                                      int n) {
  for (int e = blockIdx.x * 256 + threadIdx.x; e < n; e += gridDim.x * 256) {
    H2U u; u.h = (_Float16)x[e];
    xh[e] = u.u;
  }
}

__global__ __launch_bounds__(128) void spa_attn_kernel(const unsigned short* __restrict__ xh,
                                                       const float* __restrict__ x,
                                                       const float* __restrict__ gamma,
                                                       float* __restrict__ out) {
  __shared__ __align__(32) unsigned short smem[17 * 1024];   // 34 KB
  unsigned short* qt      = smem;              // [64][64]  queries [n][c]          (8 KB)
  unsigned short* tile_mc = smem + 4096;       // [64][64]  keys    [m][c]          (8 KB)
  unsigned short* tile_cm = smem + 8192;       // [80][64]  keys+ones [c][m]        (10 KB)
  float*          ostage  = (float*)smem;      // [64][64] f32, reused after loop   (16 KB)

  const int tid  = threadIdx.x;
  const int lane = tid & 31;
  const int wave = tid >> 5;
  const int half = lane >> 4;
  const int col  = lane & 15;

  const int b  = blockIdx.x >> 6;
  const int n0 = (blockIdx.x & 63) * QT;
  const unsigned short* xhb = xh + (size_t)b * C_DIM * N_DIM;
  const float*          xb  = x  + (size_t)b * C_DIM * N_DIM;
  unsigned short* pbuf = smem + 13312 + wave * 16 * 64;  // per-wave P^T tile [n 16][m 64]

  // Per-thread staging decomposition: 4 chunks of 8 contiguous elements, constant offsets.
  const int c0  = tid >> 3;        // 0..15   (+16 per unrolled step)
  const int mb0 = (tid & 7) * 8;   // 0..56   (fixed per thread)
  const unsigned short* gK = xhb + c0 * N_DIM + mb0;        // + m0 per iteration
  const unsigned short* gQ = xhb + c0 * N_DIM + n0 + mb0;
  unsigned short* cm0 = tile_cm + c0 * 64 + mb0;
  unsigned short* mc0 = tile_mc + mb0 * 64 + c0;

  // ---- one-time: ones row (c=64) + zero rows (65..79) of tile_cm for the row-sum GEMM ----
#pragma unroll
  for (int i = 0; i < 8; ++i) {
    int q = tid + i * 128;
    int r = 64 + (q >> 6), cc = q & 63;
    tile_cm[r * 64 + cc] = (r == 64) ? (unsigned short)0x3C00 : (unsigned short)0;
  }

  // ---- stage Q tile: qt[n][c] = Xh[c][n0+n] ----
#pragma unroll
  for (int i = 0; i < 4; ++i) {
    v4u w = *(const v4u*)(gQ + i * (16 * N_DIM));
#pragma unroll
    for (int j = 0; j < 8; ++j) {
      unsigned short u = (unsigned short)((w[j >> 1] >> (16 * (j & 1))) & 0xffffu);
      qt[(mb0 + j) * 64 + c0 + i * 16] = u;
    }
  }
  __syncthreads();

  // Query B-fragments are loop-invariant (this wave's 16 query columns).
  const v16h bq0 = load_b_frag(qt, 64, wave * 16, 0,  lane);
  const v16h bq1 = load_b_frag(qt, 64, wave * 16, 32, lane);

  v8f acc[5];   // O' tiles rows c = f*16+v+8*half, col = this lane's query; acc[4] row 64 = row-sum
#pragma unroll
  for (int f = 0; f < 5; ++f)
#pragma unroll
    for (int v = 0; v < 8; ++v) acc[f][v] = 0.0f;
  float mrun = -__builtin_inff();   // per-lane query running max (replicated in lane^16)

  for (int m0 = 0; m0 < N_DIM; m0 += MT) {
    __syncthreads();   // all waves done reading previous key tile
#pragma unroll
    for (int i = 0; i < 4; ++i) {
      v4u w = *(const v4u*)(gK + m0 + i * (16 * N_DIM));
      *(v4u*)(cm0 + i * (16 * 64)) = w;
#pragma unroll
      for (int j = 0; j < 8; ++j) {
        unsigned short u = (unsigned short)((w[j >> 1] >> (16 * (j & 1))) & 0xffffu);
        mc0[j * 64 + i * 16] = u;
      }
    }
    __syncthreads();

    // ---- S^T tiles: rows = keys, cols = queries ----
    v8f st[4];
#pragma unroll
    for (int t = 0; t < 4; ++t) {
      v16h a0 = load_a_frag(tile_mc, 64, t * 16, 0,  lane);
      v16h a1 = load_a_frag(tile_mc, 64, t * 16, 32, lane);
      v8f s = {0.f, 0.f, 0.f, 0.f, 0.f, 0.f, 0.f, 0.f};
      s = WMMA_F16(a0, bq0, s);
      s = WMMA_F16(a1, bq1, s);
      st[t] = s;
    }

    // ---- online softmax: key-row is in-lane + one xor-16; rescale only when max moves ----
    float ml = st[0][0];
#pragma unroll
    for (int t = 0; t < 4; ++t)
#pragma unroll
      for (int v = 0; v < 8; ++v) ml = fmaxf(ml, st[t][v]);
    ml = fmaxf(ml, __shfl_xor(ml, 16, 32));

    if (__any(ml > mrun)) {                       // wave-uniform branch
      float mn = fmaxf(mrun, ml);
      float sc = EXP2F((mrun - mn) * LOG2E);
#pragma unroll
      for (int f = 0; f < 5; ++f)
#pragma unroll
        for (int v = 0; v < 8; ++v) acc[f][v] *= sc;
      mrun = mn;
    }

    // P = exp(S - mrun); pack pairs and store one b128 per tile (8 consecutive m per lane)
    const float mb2 = mrun * LOG2E;
#pragma unroll
    for (int t = 0; t < 4; ++t) {
      float p[8];
#pragma unroll
      for (int v = 0; v < 8; ++v)
        p[v] = EXP2F(__builtin_fmaf(st[t][v], LOG2E, -mb2));
      v4u pk;
#pragma unroll
      for (int j = 0; j < 4; ++j) {
        P2U u; u.h = __builtin_amdgcn_cvt_pkrtz(p[2 * j], p[2 * j + 1]);
        pk[j] = u.u;
      }
      *(v4u*)(pbuf + col * 64 + t * 16 + 8 * half) = pk;
    }
    asm volatile("s_wait_dscnt 0x0" ::: "memory");

    // ---- O' += [V; ones] * P^T : 5 A tiles from tile_cm, B = P^T from pbuf ----
    v16h b20 = load_b_frag(pbuf, 64, 0, 0,  lane);
    v16h b21 = load_b_frag(pbuf, 64, 0, 32, lane);
#pragma unroll
    for (int f = 0; f < 5; ++f) {
      v16h a20 = load_a_frag(tile_cm, 64, f * 16, 0,  lane);
      v16h a21 = load_a_frag(tile_cm, 64, f * 16, 32, lane);
      acc[f] = WMMA_F16(a20, b20, acc[f]);
      acc[f] = WMMA_F16(a21, b21, acc[f]);
    }
  }
  __syncthreads();   // done with key tiles; smem reused as ostage

  // ---- softmax denominator came out of the matrix pipe: acc[4] row 64 ----
  float lsum = acc[4][0];
  lsum += __shfl_xor(lsum, 16, 32);   // transport to the h=1 half (adds that half's zero row)
  float inv = 1.0f / lsum;
#pragma unroll
  for (int f = 0; f < 4; ++f)
#pragma unroll
    for (int v = 0; v < 8; ++v) {
      int c = f * 16 + v + 8 * half;
      ostage[c * 64 + wave * 16 + col] = acc[f][v] * inv;
    }
  __syncthreads();

  const float g = gamma[0];
  float* ob = out + (size_t)b * C_DIM * N_DIM;
#pragma unroll 4
  for (int e = tid; e < QT * C_DIM; e += 128) {
    int c = e >> 6, n = e & 63;
    int gi = c * N_DIM + n0 + n;
    ob[gi] = g * ostage[c * 64 + n] + xb[gi];
  }
}

extern "C" void kernel_launch(void* const* d_in, const int* in_sizes, int n_in,
                              void* d_out, int out_size, void* d_ws, size_t ws_size,
                              hipStream_t stream) {
  const float* x     = (const float*)d_in[0];
  const float* gamma = (const float*)d_in[1];
  float* out = (float*)d_out;
  const int total = 4 * C_DIM * N_DIM;              // 1,048,576 elements
  unsigned short* xh = (unsigned short*)d_ws;       // needs 2 MB scratch (f16 copy of x)

  hipLaunchKernelGGL(cvt_f16_kernel, dim3(512), dim3(256), 0, stream, x, xh, total);
  hipLaunchKernelGGL(spa_attn_kernel, dim3(4 * (N_DIM / QT)), dim3(128), 0, stream,
                     xh, x, gamma, out);
}
